// CausalSelfAttention_12360915878502
// MI455X (gfx1250) — matrix operand order repaired and checked by reading, not compiled, not run on hardware
//
#include <hip/hip_runtime.h>

#define B_N   2
#define T_SEQ 2048
#define C_DIM 1024
#define H_N   16
#define D_H   64
#define N_QKV (3 * C_DIM)

typedef __attribute__((ext_vector_type(16))) _Float16 v16h;
typedef __attribute__((ext_vector_type(8)))  _Float16 v8h;
typedef __attribute__((ext_vector_type(4)))  _Float16 v4h;
typedef __attribute__((ext_vector_type(8)))  float    v8f;

// ---------------------------------------------------------------------------
// CDNA5 async global->LDS copy (16B per lane), tracked by ASYNCcnt.
// ---------------------------------------------------------------------------
__device__ __forceinline__ void async_copy_b128(void* lds_dst, const void* gsrc) {
    const unsigned lds_addr = (unsigned)(size_t)lds_dst;
    asm volatile("global_load_async_to_lds_b128 %0, %1, off"
                 :: "v"(lds_addr), "v"(gsrc)
                 : "memory");
}
__device__ __forceinline__ void async_wait0() {
    asm volatile("s_wait_asynccnt 0x0" ::: "memory");
}

// ---------------------------------------------------------------------------
// Wide WMMA fragment loaders (2x ds_load_b128 per fragment).
// A fragment from row-major [m][k] tile; B fragment from n-major [n][k] tile.
// ---------------------------------------------------------------------------
__device__ __forceinline__ v16h lds_load_a(const _Float16* __restrict__ src, int ld) {
    const int lane = threadIdx.x & 31;
    const int m  = lane & 15;
    const int hi = lane >> 4;
    const v8h lo = *(const v8h*)&src[m * ld + 8 * hi];
    const v8h hh = *(const v8h*)&src[m * ld + 16 + 8 * hi];
    return __builtin_shufflevector(lo, hh, 0,1,2,3,4,5,6,7,8,9,10,11,12,13,14,15);
}

__device__ __forceinline__ v16h lds_load_b(const _Float16* __restrict__ src, int ld) {
    const int lane = threadIdx.x & 31;
    const int n  = lane & 15;
    const int hi = lane >> 4;
    const v8h lo = *(const v8h*)&src[n * ld + 8 * hi];
    const v8h hh = *(const v8h*)&src[n * ld + 16 + 8 * hi];
    return __builtin_shufflevector(lo, hh, 0,1,2,3,4,5,6,7,8,9,10,11,12,13,14,15);
}

#define WMMA(A, B, C) __builtin_amdgcn_wmma_f32_16x16x32_f16(false, A, false, B, (short)0, C, false, false)

// ---------------------------------------------------------------------------
// Preprocess: x (f32) -> f16, straight copy.
// ---------------------------------------------------------------------------
__global__ __launch_bounds__(256)
void convert_f32_f16_kernel(const float* __restrict__ src, _Float16* __restrict__ dst) {
    const size_t i = ((size_t)blockIdx.x * 256 + threadIdx.x) * 8;
    const float4 a = *(const float4*)&src[i];
    const float4 b = *(const float4*)&src[i + 4];
    v8h o;
    o[0] = (_Float16)a.x; o[1] = (_Float16)a.y; o[2] = (_Float16)a.z; o[3] = (_Float16)a.w;
    o[4] = (_Float16)b.x; o[5] = (_Float16)b.y; o[6] = (_Float16)b.z; o[7] = (_Float16)b.w;
    *(v8h*)&dst[i] = o;
}

// ---------------------------------------------------------------------------
// Preprocess: W [rows=K][cols=N] f32 -> WT [N][K] f16 (32x32 LDS tiles).
// ---------------------------------------------------------------------------
__global__ __launch_bounds__(256)
void transpose_f32_f16_kernel(const float* __restrict__ src, _Float16* __restrict__ dst,
                              int rows, int cols) {
    __shared__ __align__(16) _Float16 tile[32][36];
    const int tid   = threadIdx.x;
    const int rBase = blockIdx.y * 32;
    const int cBase = blockIdx.x * 32;
    const int r  = tid >> 3;
    const int c4 = (tid & 7) * 4;
    const float4 v = *(const float4*)&src[(size_t)(rBase + r) * cols + cBase + c4];
    tile[c4 + 0][r] = (_Float16)v.x;
    tile[c4 + 1][r] = (_Float16)v.y;
    tile[c4 + 2][r] = (_Float16)v.z;
    tile[c4 + 3][r] = (_Float16)v.w;
    __syncthreads();
    const int n  = tid >> 3;
    const int k4 = (tid & 7) * 4;
    *(v4h*)&dst[(size_t)(cBase + n) * rows + rBase + k4] = *(const v4h*)&tile[n][k4];
}

// ---------------------------------------------------------------------------
// Kernel 1: qkv = Xh @ WqkvT^T. Block tile 64x256, 8 waves each owning a
// 32x64 C tile. K-chunks of 64, double-buffered LDS with async DMA overlapped
// under 16 WMMAs/chunk. Writes Q (x0.125), K in [B,H,T,D]; V into [B,H,D,T].
// ---------------------------------------------------------------------------
__global__ __launch_bounds__(256)
void qkv_gemm_kernel(const _Float16* __restrict__ Xh, const _Float16* __restrict__ WT,
                     _Float16* __restrict__ Q, _Float16* __restrict__ K,
                     _Float16* __restrict__ Vt) {
    __shared__ __align__(16) _Float16 As[2][64][64];    // [buf][m][k]
    __shared__ __align__(16) _Float16 Bs[2][256][64];   // [buf][n][k]

    const int tid  = threadIdx.x;
    const int wave = tid >> 5;
    const int lane = tid & 31;
    const int hi   = lane >> 4;
    const int ncol = lane & 15;
    const int rowBase = blockIdx.y * 64;
    const int colBase = blockIdx.x * 256;
    const int rowSub  = (wave & 1) * 32;
    const int colSub  = (wave >> 1) * 64;

    auto stage = [&](int buf, int kk) {
#pragma unroll
        for (int j = 0; j < 2; ++j) {                   // A: 64x64 = 512 x 16B
            const int v  = tid + j * 256;
            const int mr = v >> 3, mk = (v & 7) * 8;
            async_copy_b128(&As[buf][mr][mk],
                            &Xh[(size_t)(rowBase + mr) * C_DIM + kk + mk]);
        }
#pragma unroll
        for (int j = 0; j < 8; ++j) {                   // B: 256x64 = 2048 x 16B
            const int v  = tid + j * 256;
            const int bn = v >> 3, bk = (v & 7) * 8;
            async_copy_b128(&Bs[buf][bn][bk],
                            &WT[(size_t)(colBase + bn) * C_DIM + kk + bk]);
        }
        if (kk + 64 < C_DIM)
            __builtin_prefetch(&WT[(size_t)(colBase + tid) * C_DIM + kk + 64], 0, 0);
    };

    v8f acc[2][4];
#pragma unroll
    for (int i = 0; i < 2; ++i)
#pragma unroll
        for (int j = 0; j < 4; ++j) acc[i][j] = (v8f){};

    stage(0, 0);
    async_wait0();
    __syncthreads();

    int cur = 0;
    for (int kk = 0; kk < C_DIM; kk += 64) {
        const int nxt = cur ^ 1;
        if (kk + 64 < C_DIM) stage(nxt, kk + 64);       // DMA overlapped with WMMAs

#pragma unroll
        for (int ks = 0; ks < 64; ks += 32) {
            v16h a[2], b[4];
#pragma unroll
            for (int i = 0; i < 2; ++i)
                a[i] = lds_load_a(&As[cur][rowSub + 16 * i][ks], 64);
#pragma unroll
            for (int j = 0; j < 4; ++j)
                b[j] = lds_load_b(&Bs[cur][colSub + 16 * j][ks], 64);
#pragma unroll
            for (int i = 0; i < 2; ++i)
#pragma unroll
                for (int j = 0; j < 4; ++j) acc[i][j] = WMMA(a[i], b[j], acc[i][j]);
        }
        async_wait0();
        __syncthreads();
        cur = nxt;
    }

#pragma unroll
    for (int j = 0; j < 4; ++j) {
        const int col   = colBase + colSub + 16 * j + ncol;
        const int which = col >> 10;                    // 0=q, 1=k, 2=v
        const int c     = col & 1023;
        const int h     = c >> 6;
        const int d     = c & 63;
        const float scale = (which == 0) ? 0.125f : 1.0f;
#pragma unroll
        for (int i = 0; i < 2; ++i) {
#pragma unroll
            for (int r = 0; r < 8; ++r) {
                const int row = rowBase + rowSub + 16 * i + r + 8 * hi;
                const int bb = row >> 11;               // T_SEQ == 2048
                const int tt = row & (T_SEQ - 1);
                const _Float16 val = (_Float16)(acc[i][j][r] * scale);
                if (which == 0)
                    Q [((size_t)(bb * H_N + h) * T_SEQ + tt) * D_H + d] = val;
                else if (which == 1)
                    K [((size_t)(bb * H_N + h) * T_SEQ + tt) * D_H + d] = val;
                else
                    Vt[((size_t)(bb * H_N + h) * D_H + d) * T_SEQ + tt] = val;
            }
        }
    }
}

// ---------------------------------------------------------------------------
// Kernel 2: causal flash attention. Block = 64 query rows of one (b,h),
// 4 waves x 16 query rows; K/V streamed in 32-key chunks, double-buffered
// async DMA overlapped with the S/softmax/PV phases.
// ---------------------------------------------------------------------------
__global__ __launch_bounds__(128)
void attn_kernel(const _Float16* __restrict__ Q, const _Float16* __restrict__ K,
                 const _Float16* __restrict__ Vt, _Float16* __restrict__ Y) {
    __shared__ __align__(16) _Float16 Qs[64][64];       // [m][d]
    __shared__ __align__(16) _Float16 Ks[2][32][64];    // [buf][key][d]
    __shared__ __align__(16) _Float16 Vs[2][64][32];    // [buf][d][key]
    __shared__ __align__(16) _Float16 Ps[4][16][32];    // per-wave P tile [m][key]

    const int tid  = threadIdx.x;
    const int wave = tid >> 5;
    const int lane = tid & 31;
    const int hi   = lane >> 4;
    const int ncol = lane & 15;
    const int bh    = blockIdx.x;
    const int qbase = blockIdx.y * 64;
    const size_t headTD = (size_t)bh * T_SEQ * D_H;
    const size_t headDT = (size_t)bh * D_H * T_SEQ;

    auto stageKV = [&](int buf, int kb) {
        const _Float16* ks = K + headTD + (size_t)kb * D_H;
#pragma unroll
        for (int j = 0; j < 2; ++j) {
            const int v = tid + j * 128;
            async_copy_b128(&Ks[buf][0][0] + v * 8, ks + v * 8);
        }
#pragma unroll
        for (int j = 0; j < 2; ++j) {
            const int v  = tid * 2 + j;
            const int d  = v >> 2;
            const int ko = (v & 3) * 8;
            async_copy_b128(&Vs[buf][d][ko], &Vt[headDT + (size_t)d * T_SEQ + kb + ko]);
        }
    };

    { // Q tile (8 KB) + first K/V chunk
        const _Float16* src = Q + headTD + (size_t)qbase * D_H;
#pragma unroll
        for (int j = 0; j < 4; ++j) {
            const int v = tid + j * 128;
            async_copy_b128(&Qs[0][0] + v * 8, src + v * 8);
        }
        stageKV(0, 0);
        async_wait0();
        __syncthreads();
    }

    float m_i[8], l_i[8];
    v8f o0 = {}, o1 = {}, o2 = {}, o3 = {};
#pragma unroll
    for (int r = 0; r < 8; ++r) { m_i[r] = -1e30f; l_i[r] = 0.0f; }

    const int q0   = qbase + wave * 16;
    const int kmax = qbase + 64;

    int cur = 0;
    for (int kb = 0; kb < kmax; kb += 32) {
        const int nxt = cur ^ 1;
        if (kb + 32 < kmax) stageKV(nxt, kb + 32);      // DMA under compute

        // S = Q K^T : 16 rows x 32 keys per wave (D=64 -> two K=32 chunks)
        const v16h aq0 = lds_load_a(&Qs[wave * 16][0],  64);
        const v16h aq1 = lds_load_a(&Qs[wave * 16][32], 64);
        v8f s0 = {}, s1 = {};
        {
            const v16h b00 = lds_load_b(&Ks[cur][0][0],   64);
            const v16h b01 = lds_load_b(&Ks[cur][0][32],  64);
            const v16h b10 = lds_load_b(&Ks[cur][16][0],  64);
            const v16h b11 = lds_load_b(&Ks[cur][16][32], 64);
            s0 = WMMA(aq0, b00, s0);
            s0 = WMMA(aq1, b01, s0);
            s1 = WMMA(aq0, b10, s1);
            s1 = WMMA(aq1, b11, s1);
        }

        // online softmax; row stats reduced across the 16-lane C-fragment group
#pragma unroll
        for (int r = 0; r < 8; ++r) {
            const int qrow = q0 + r + 8 * hi;
            float a0 = s0[r];
            float a1 = s1[r];
            if (kb + ncol      > qrow) a0 = -1e30f;
            if (kb + 16 + ncol > qrow) a1 = -1e30f;
            float mx = fmaxf(a0, a1);
            mx = fmaxf(mx, __shfl_xor(mx, 1, 32));
            mx = fmaxf(mx, __shfl_xor(mx, 2, 32));
            mx = fmaxf(mx, __shfl_xor(mx, 4, 32));
            mx = fmaxf(mx, __shfl_xor(mx, 8, 32));
            const float mn = fmaxf(m_i[r], mx);
            const float p0 = __expf(a0 - mn);
            const float p1 = __expf(a1 - mn);
            float rs = p0 + p1;
            rs += __shfl_xor(rs, 1, 32);
            rs += __shfl_xor(rs, 2, 32);
            rs += __shfl_xor(rs, 4, 32);
            rs += __shfl_xor(rs, 8, 32);
            const float corr = __expf(m_i[r] - mn);
            l_i[r] = l_i[r] * corr + rs;
            m_i[r] = mn;
            o0[r] *= corr; o1[r] *= corr; o2[r] *= corr; o3[r] *= corr;
            Ps[wave][r + 8 * hi][ncol]      = (_Float16)p0;
            Ps[wave][r + 8 * hi][16 + ncol] = (_Float16)p1;
        }
        __syncthreads();                                // publish Ps

        // O += P @ V
        const v16h ap  = lds_load_a(&Ps[wave][0][0], 32);
        const v16h bv0 = lds_load_b(&Vs[cur][0][0],  32);
        const v16h bv1 = lds_load_b(&Vs[cur][16][0], 32);
        const v16h bv2 = lds_load_b(&Vs[cur][32][0], 32);
        const v16h bv3 = lds_load_b(&Vs[cur][48][0], 32);
        o0 = WMMA(ap, bv0, o0);
        o1 = WMMA(ap, bv1, o1);
        o2 = WMMA(ap, bv2, o2);
        o3 = WMMA(ap, bv3, o3);

        async_wait0();
        __syncthreads();
        cur = nxt;
    }

    const int b = bh >> 4;
    const int h = bh & 15;
#pragma unroll
    for (int r = 0; r < 8; ++r) {
        const float inv  = 1.0f / l_i[r];
        const int   qrow = q0 + r + 8 * hi;
        _Float16* yrow = Y + (size_t)(b * T_SEQ + qrow) * C_DIM + h * D_H;
        yrow[ncol]      = (_Float16)(o0[r] * inv);
        yrow[16 + ncol] = (_Float16)(o1[r] * inv);
        yrow[32 + ncol] = (_Float16)(o2[r] * inv);
        yrow[48 + ncol] = (_Float16)(o3[r] * inv);
    }
}

// ---------------------------------------------------------------------------
// Kernel 3: out = Y @ WprojT^T, fp32 output. Same pipeline as kernel 1.
// ---------------------------------------------------------------------------
__global__ __launch_bounds__(256)
void proj_gemm_kernel(const _Float16* __restrict__ Yh, const _Float16* __restrict__ WT,
                      float* __restrict__ out) {
    __shared__ __align__(16) _Float16 As[2][64][64];
    __shared__ __align__(16) _Float16 Bs[2][256][64];

    const int tid  = threadIdx.x;
    const int wave = tid >> 5;
    const int lane = tid & 31;
    const int hi   = lane >> 4;
    const int ncol = lane & 15;
    const int rowBase = blockIdx.y * 64;
    const int colBase = blockIdx.x * 256;
    const int rowSub  = (wave & 1) * 32;
    const int colSub  = (wave >> 1) * 64;

    auto stage = [&](int buf, int kk) {
#pragma unroll
        for (int j = 0; j < 2; ++j) {
            const int v  = tid + j * 256;
            const int mr = v >> 3, mk = (v & 7) * 8;
            async_copy_b128(&As[buf][mr][mk],
                            &Yh[(size_t)(rowBase + mr) * C_DIM + kk + mk]);
        }
#pragma unroll
        for (int j = 0; j < 8; ++j) {
            const int v  = tid + j * 256;
            const int bn = v >> 3, bk = (v & 7) * 8;
            async_copy_b128(&Bs[buf][bn][bk],
                            &WT[(size_t)(colBase + bn) * C_DIM + kk + bk]);
        }
        if (kk + 64 < C_DIM)
            __builtin_prefetch(&WT[(size_t)(colBase + tid) * C_DIM + kk + 64], 0, 0);
    };

    v8f acc[2][4];
#pragma unroll
    for (int i = 0; i < 2; ++i)
#pragma unroll
        for (int j = 0; j < 4; ++j) acc[i][j] = (v8f){};

    stage(0, 0);
    async_wait0();
    __syncthreads();

    int cur = 0;
    for (int kk = 0; kk < C_DIM; kk += 64) {
        const int nxt = cur ^ 1;
        if (kk + 64 < C_DIM) stage(nxt, kk + 64);

#pragma unroll
        for (int ks = 0; ks < 64; ks += 32) {
            v16h a[2], b[4];
#pragma unroll
            for (int i = 0; i < 2; ++i)
                a[i] = lds_load_a(&As[cur][rowSub + 16 * i][ks], 64);
#pragma unroll
            for (int j = 0; j < 4; ++j)
                b[j] = lds_load_b(&Bs[cur][colSub + 16 * j][ks], 64);
#pragma unroll
            for (int i = 0; i < 2; ++i)
#pragma unroll
                for (int j = 0; j < 4; ++j) acc[i][j] = WMMA(a[i], b[j], acc[i][j]);
        }
        async_wait0();
        __syncthreads();
        cur = nxt;
    }

#pragma unroll
    for (int j = 0; j < 4; ++j) {
        const int col = colBase + colSub + 16 * j + ncol;
#pragma unroll
        for (int i = 0; i < 2; ++i) {
#pragma unroll
            for (int r = 0; r < 8; ++r) {
                const int row = rowBase + rowSub + 16 * i + r + 8 * hi;
                out[(size_t)row * C_DIM + col] = acc[i][j][r];
            }
        }
    }
}

// ---------------------------------------------------------------------------
extern "C" void kernel_launch(void* const* d_in, const int* in_sizes, int n_in,
                              void* d_out, int out_size, void* d_ws, size_t ws_size,
                              hipStream_t stream) {
    (void)in_sizes; (void)n_in; (void)out_size; (void)ws_size;
    const float* x     = (const float*)d_in[0];
    const float* Wqkv  = (const float*)d_in[1];
    const float* Wproj = (const float*)d_in[2];
    float* out = (float*)d_out;

    const size_t headElems = (size_t)B_N * H_N * T_SEQ * D_H;  // 4,194,304
    _Float16* Qws    = (_Float16*)d_ws;
    _Float16* Kws    = Qws    + headElems;
    _Float16* Vtws   = Kws    + headElems;
    _Float16* Yws    = Vtws   + headElems;
    _Float16* Xh     = Yws    + headElems;
    _Float16* WqkvT  = Xh     + (size_t)B_N * T_SEQ * C_DIM;
    _Float16* WprojT = WqkvT  + (size_t)N_QKV * C_DIM;         // ~50 MB total

    convert_f32_f16_kernel<<<dim3((B_N * T_SEQ * C_DIM) / (256 * 8)), 256, 0, stream>>>(x, Xh);
    transpose_f32_f16_kernel<<<dim3(N_QKV / 32, C_DIM / 32), 256, 0, stream>>>(Wqkv, WqkvT, C_DIM, N_QKV);
    transpose_f32_f16_kernel<<<dim3(C_DIM / 32, C_DIM / 32), 256, 0, stream>>>(Wproj, WprojT, C_DIM, C_DIM);

    dim3 g1(N_QKV / 256, (B_N * T_SEQ) / 64);                  // 12 x 64
    qkv_gemm_kernel<<<g1, 256, 0, stream>>>(Xh, WqkvT, Qws, Kws, Vtws);

    dim3 g2(B_N * H_N, T_SEQ / 64);                            // 32 x 32
    attn_kernel<<<g2, 128, 0, stream>>>(Qws, Kws, Vtws, Yws);

    dim3 g3(C_DIM / 256, (B_N * T_SEQ) / 64);                  // 4 x 64
    proj_gemm_kernel<<<g3, 256, 0, stream>>>(Yws, WprojT, out);
}
